// LDPCDecoder_40346922779484
// MI455X (gfx1250) — 8-bit matrix operand order repaired and checked by reading, compiled, not run on hardware
//
#include <hip/hip_runtime.h>
#include <hip/hip_bf16.h>
#include <stdint.h>


typedef __attribute__((ext_vector_type(2))) float v2f;
typedef __attribute__((ext_vector_type(4))) float v4f;
typedef __attribute__((ext_vector_type(8))) float v8f;
typedef __attribute__((ext_vector_type(8))) int   v8i;

#define NCOL 64
#define ITERS 10
#define TPB 256
#define WPB 8
#define BLOCKS 512
#define ASTRIDE 68
#define SSTRIDE 72


__device__ __forceinline__ float llr_of(float x) { return 2.0f * x / (1.0f - x); }

__device__ __forceinline__ void async_g2l_b128(uint32_t lds_byte_addr, const void* gaddr) {
  asm volatile("global_load_async_to_lds_b128 %0, %1, off"
               :: "v"(lds_byte_addr), "v"((uint64_t)(uintptr_t)gaddr)
               : "memory");
}
__device__ __forceinline__ void wait_asynccnt0() {
  asm volatile("s_wait_asynccnt 0x0" ::: "memory");
}

__global__ void ldpc_prep(const float* __restrict__ H, float* __restrict__ Gfrag,
                          signed char* __restrict__ Hbfrag, int* __restrict__ flags) {
  int t = threadIdx.x;
  for (int idx = t; idx < 4096; idx += TPB) {
    int j  = idx & 1;
    int ln = (idx >> 1) & 31;
    int nt = (idx >> 6) & 3;
    int ks = idx >> 8;
    int K  = 4 * ks + ((ln < 16) ? j : (j + 2));
    int Nc = 16 * nt + (ln & 15);
    Gfrag[idx] = H[Nc * NCOL + K] + ((K == Nc) ? 1.0f : 0.0f);
  }
  for (int idx = t; idx < 4096; idx += TPB) {
    int b  = idx & 31;
    int ln = (idx >> 5) & 31;
    int nt = idx >> 10;
    int base = (ln < 16) ? 0 : 16;
    int K = (b >> 3) * 16 + (base >> 1) + (b & 7);
    int Nc = 16 * nt + (ln & 15);
    Hbfrag[idx] = (signed char)H[Nc * NCOL + K];
  }
  if (t < ITERS) flags[t] = 0;
}

__global__ __launch_bounds__(TPB) void ldpc_pass1(
    const float* __restrict__ r, const float* __restrict__ Gfrag,
    const signed char* __restrict__ Hbfrag, int* __restrict__ flags,
    int* __restrict__ out, int nTiles) {
  __shared__ float ldsG[4096];
  __shared__ float ldsA[WPB][16 * ASTRIDE];
  __shared__ signed char ldsS[WPB][16 * SSTRIDE];
  __shared__ int bflags;

  const int tid = threadIdx.x;
  for (int i = tid; i < 4096; i += TPB) ldsG[i] = Gfrag[i];
  if (tid == 0) bflags = 0;
  __syncthreads();

  const int lane = tid & 31;
  const int wave = tid >> 5;
  float* As = ldsA[wave];
  signed char* Ss = ldsS[wave];
  const uint32_t AsBase = (uint32_t)(uintptr_t)(void*)As;

  v8i hb[4];
#pragma unroll
  for (int nt = 0; nt < 4; ++nt)
    hb[nt] = *(const v8i*)(Hbfrag + (size_t)(nt * 32 + lane) * 32);

  const int gw = blockIdx.x * WPB + wave;
  const int nWaves = gridDim.x * WPB;
  const int r0 = lane & 15;
  const int koff = (lane < 16) ? 0 : 2;
  const int so = (lane < 16) ? 0 : 8;
  const int mb = (lane < 16) ? 0 : 8;

  int oddbits = 0;

  for (int tile = gw; tile < nTiles; tile += nWaves) {
    const char* gbase = (const char*)(r + (size_t)tile * (16 * NCOL));
#pragma unroll
    for (int i = 0; i < 8; ++i) {
      int f = lane + 32 * i;
      int row = f >> 4;
      int colb = (f & 15) * 16;
      async_g2l_b128(AsBase + (uint32_t)(row * (ASTRIDE * 4) + colb),
                     gbase + (size_t)f * 16);
    }
    wait_asynccnt0();

    v2f a[16];
#pragma unroll
    for (int ks = 0; ks < 16; ++ks) {
      v2f raw = *(const v2f*)&As[r0 * ASTRIDE + 4 * ks + koff];
      a[ks] = v2f{llr_of(raw.x), llr_of(raw.y)};
    }

    for (int it = 0; it < ITERS; ++it) {
      v8f acc[4];
#pragma unroll
      for (int nt = 0; nt < 4; ++nt)
        acc[nt] = v8f{0.f, 0.f, 0.f, 0.f, 0.f, 0.f, 0.f, 0.f};
#pragma unroll
      for (int ks = 0; ks < 16; ++ks) {
#pragma unroll
        for (int nt = 0; nt < 4; ++nt) {
          v2f bg = *(const v2f*)&ldsG[(((ks << 2) + nt) * 32 + lane) * 2];
          acc[nt] = __builtin_amdgcn_wmma_f32_16x16x4_f32(
              false, a[ks], false, bg, (short)0, acc[nt], false, false);
        }
      }
#pragma unroll
      for (int nt = 0; nt < 4; ++nt) {
#pragma unroll
        for (int v = 0; v < 8; ++v) {
          float x = acc[nt][v];
          int s = (x > 0.0f) - (x < 0.0f);
          int M = mb + v;
          int col = nt * 16 + r0;
          Ss[M * SSTRIDE + col] = (signed char)s;
          As[M * ASTRIDE + col] = x;
          if (it == ITERS - 1)
            out[((size_t)tile * 16 + M) * NCOL + col] = s;
        }
      }
#pragma unroll
      for (int ks = 0; ks < 16; ++ks)
        a[ks] = *(const v2f*)&As[r0 * ASTRIDE + 4 * ks + koff];
      v8i dA;
#pragma unroll
      for (int g = 0; g < 4; ++g) {
        uint64_t q = *(const uint64_t*)(Ss + r0 * SSTRIDE + g * 16 + so);
        dA[2 * g + 0] = (int)(uint32_t)q;
        dA[2 * g + 1] = (int)(uint32_t)(q >> 32);
      }
      const v8i zero = v8i{0, 0, 0, 0, 0, 0, 0, 0};
      int odd = 0;
#pragma unroll
      for (int nt = 0; nt < 4; ++nt) {
        v8i syn = __builtin_amdgcn_wmma_i32_16x16x64_iu8(
            true, dA, false, hb[nt], zero, false, false);
#pragma unroll
        for (int v = 0; v < 8; ++v) odd |= (syn[v] & 1);
      }
      if (odd) oddbits |= (1 << it);
    }
  }
  atomicOr(&bflags, oddbits);
  __syncthreads();
  if (tid == 0 && bflags) {
    int bf = bflags;
    for (int i = 0; i < ITERS; ++i)
      if (bf & (1 << i)) atomicOr(&flags[i], 1);
  }
}

__global__ void ldpc_select(const int* __restrict__ flags, int* __restrict__ ksel) {
  int k = ITERS - 1;
  for (int i = 0; i < ITERS; ++i) {
    if (flags[i] == 0) { k = i; break; }
  }
  ksel[0] = k;
}

__global__ __launch_bounds__(TPB) void ldpc_pass2(
    const float* __restrict__ r, const float* __restrict__ Gfrag,
    const int* __restrict__ ksel, int* __restrict__ out, int nTiles) {
  const int k = ksel[0];
  if (k >= ITERS - 1) return;
  __shared__ float ldsG[4096];
  __shared__ float ldsA[WPB][16 * ASTRIDE];

  const int tid = threadIdx.x;
  for (int i = tid; i < 4096; i += TPB) ldsG[i] = Gfrag[i];
  __syncthreads();

  const int lane = tid & 31;
  const int wave = tid >> 5;
  float* As = ldsA[wave];
  const uint32_t AsBase = (uint32_t)(uintptr_t)(void*)As;
  const int gw = blockIdx.x * WPB + wave;
  const int nWaves = gridDim.x * WPB;
  const int r0 = lane & 15;
  const int koff = (lane < 16) ? 0 : 2;
  const int mb = (lane < 16) ? 0 : 8;

  for (int tile = gw; tile < nTiles; tile += nWaves) {
    const char* gbase = (const char*)(r + (size_t)tile * (16 * NCOL));
#pragma unroll
    for (int i = 0; i < 8; ++i) {
      int f = lane + 32 * i;
      int row = f >> 4;
      int colb = (f & 15) * 16;
      async_g2l_b128(AsBase + (uint32_t)(row * (ASTRIDE * 4) + colb),
                     gbase + (size_t)f * 16);
    }
    wait_asynccnt0();

    v2f a[16];
#pragma unroll
    for (int ks = 0; ks < 16; ++ks) {
      v2f raw = *(const v2f*)&As[r0 * ASTRIDE + 4 * ks + koff];
      a[ks] = v2f{llr_of(raw.x), llr_of(raw.y)};
    }

    for (int it = 0; it <= k; ++it) {
      v8f acc[4];
#pragma unroll
      for (int nt = 0; nt < 4; ++nt)
        acc[nt] = v8f{0.f, 0.f, 0.f, 0.f, 0.f, 0.f, 0.f, 0.f};
#pragma unroll
      for (int ks = 0; ks < 16; ++ks) {
#pragma unroll
        for (int nt = 0; nt < 4; ++nt) {
          v2f bg = *(const v2f*)&ldsG[(((ks << 2) + nt) * 32 + lane) * 2];
          acc[nt] = __builtin_amdgcn_wmma_f32_16x16x4_f32(
              false, a[ks], false, bg, (short)0, acc[nt], false, false);
        }
      }
      if (it == k) {
#pragma unroll
        for (int nt = 0; nt < 4; ++nt) {
#pragma unroll
          for (int v = 0; v < 8; ++v) {
            float x = acc[nt][v];
            int s = (x > 0.0f) - (x < 0.0f);
            out[((size_t)tile * 16 + mb + v) * NCOL + nt * 16 + r0] = s;
          }
        }
      } else {
#pragma unroll
        for (int nt = 0; nt < 4; ++nt) {
#pragma unroll
          for (int v = 0; v < 8; ++v)
            As[(mb + v) * ASTRIDE + nt * 16 + r0] = acc[nt][v];
        }
#pragma unroll
        for (int ks = 0; ks < 16; ++ks)
          a[ks] = *(const v2f*)&As[r0 * ASTRIDE + 4 * ks + koff];
      }
    }
  }
}

extern "C" void kernel_launch(void* const* d_in, const int* in_sizes, int n_in,
                              void* d_out, int out_size, void* d_ws, size_t ws_size,
                              hipStream_t stream) {
  const float* r = (const float*)d_in[0];
  const float* H = (const float*)d_in[1];
  int* out = (int*)d_out;
  char* ws = (char*)d_ws;
  float* Gfrag = (float*)ws;
  signed char* Hbfrag = (signed char*)(ws + 16384);
  int* flags = (int*)(ws + 20480);
  int* ksel = (int*)(ws + 20544);
  int nTiles = in_sizes[0] / (16 * NCOL);

  ldpc_prep<<<1, TPB, 0, stream>>>(H, Gfrag, Hbfrag, flags);
  ldpc_pass1<<<BLOCKS, TPB, 0, stream>>>(r, Gfrag, Hbfrag, flags, out, nTiles);
  ldpc_select<<<1, 1, 0, stream>>>(flags, ksel);
  ldpc_pass2<<<BLOCKS, TPB, 0, stream>>>(r, Gfrag, ksel, out, nTiles);
}
